// mLSTMCell_7224134992402
// MI455X (gfx1250) — matrix operand order repaired and checked by reading, not compiled, not run on hardware
//
#include <hip/hip_runtime.h>
#include <hip/hip_bf16.h>
#include <math.h>

// mLSTM forward for MI455X (gfx1250): flash-style tiling, bf16 WMMA matmuls,
// separable row-stabilizer (m_i = csum_i + prefixmax(g)_i) so no online rescale.
// K pre-converted to bf16 (row major), V pre-transposed to bf16 [NH][DH][S] so
// every WMMA B operand is a directly-loadable aligned v16bf.

#define SEQ 2048
#define EMB 1024
#define NHEAD 8
#define DHEAD 128
#define EPS_N 1e-6f
#define LN_EPS 1e-5f

typedef __attribute__((ext_vector_type(16))) __bf16 v16bf;
typedef __attribute__((ext_vector_type(8)))  float  v8f;

// ---------------- Kernel 1: gate projections ig, logsigmoid(fg) ------------
__global__ void __launch_bounds__(256)
gate_kernel(const float* __restrict__ q, const float* __restrict__ k,
            const float* __restrict__ v, const float* __restrict__ igw,
            const float* __restrict__ igb, const float* __restrict__ fgw,
            const float* __restrict__ fgb,
            float* __restrict__ igt, float* __restrict__ lft) {
  const int s = blockIdx.x;
  const int h = threadIdx.x >> 5;
  const int lane = threadIdx.x & 31;
  const float* wi = igw + h * 3 * EMB;
  const float* wf = fgw + h * 3 * EMB;
  float ai = 0.f, af = 0.f;
  for (int j = lane; j < EMB; j += 32) {
    float xq = q[(size_t)s * EMB + j];
    float xk = k[(size_t)s * EMB + j];
    float xv = v[(size_t)s * EMB + j];
    ai += xq * wi[j] + xk * wi[EMB + j] + xv * wi[2 * EMB + j];
    af += xq * wf[j] + xk * wf[EMB + j] + xv * wf[2 * EMB + j];
  }
#pragma unroll
  for (int m = 16; m >= 1; m >>= 1) {
    ai += __shfl_xor(ai, m, 32);
    af += __shfl_xor(af, m, 32);
  }
  if (lane == 0) {
    float ig = ai + igb[h];
    float fp = af + fgb[h];
    float ls = fminf(fp, 0.f) - log1pf(expf(-fabsf(fp)));  // log_sigmoid
    igt[h * SEQ + s] = ig;
    lft[h * SEQ + s] = ls;
  }
}

// ---------------- Kernel 2: per-head scans ---------------------------------
// csum = inclusive cumsum(logf); g = ig - csum; gmax = prefix-max(g);
// m = csum + gmax.   One wave per head, 64 elements per lane.
__global__ void __launch_bounds__(32)
scan_kernel(const float* __restrict__ igt, const float* __restrict__ lft,
            float* __restrict__ gb, float* __restrict__ gmx,
            float* __restrict__ mb) {
  const int h = blockIdx.x;
  const int lane = threadIdx.x;
  const int CH = SEQ / 32;
  const int base = h * SEQ + lane * CH;
  float loc = 0.f;
  for (int t = 0; t < CH; ++t) loc += lft[base + t];
  float inc = loc;
#pragma unroll
  for (int o = 1; o < 32; o <<= 1) {
    float n = __shfl_up(inc, o, 32);
    if (lane >= o) inc += n;
  }
  float run = inc - loc;  // exclusive prefix sum
  float lmax = -INFINITY;
  for (int t = 0; t < CH; ++t) {
    run += lft[base + t];
    float g = igt[base + t] - run;
    gb[base + t] = g;
    mb[base + t] = run;  // stash csum
    lmax = fmaxf(lmax, g);
  }
  float incm = lmax;
#pragma unroll
  for (int o = 1; o < 32; o <<= 1) {
    float n = __shfl_up(incm, o, 32);
    if (lane >= o) incm = fmaxf(incm, n);
  }
  float em = __shfl_up(incm, 1, 32);
  float rmax = (lane == 0) ? -INFINITY : em;  // exclusive prefix max
  for (int t = 0; t < CH; ++t) {
    rmax = fmaxf(rmax, gb[base + t]);
    gmx[base + t] = rmax;
    mb[base + t] += rmax;  // m = csum + gmax
  }
}

// ---------------- Kernel 2b: K -> bf16, [NH][S][DH] row-major --------------
__global__ void __launch_bounds__(256)
kconv_kernel(const float* __restrict__ k, __bf16* __restrict__ kbf) {
  const int s = blockIdx.x;
  const int t = threadIdx.x;
#pragma unroll
  for (int i = 0; i < 4; ++i) {
    int e = t * 4 + i;            // 0..1023; head block of 128 never split
    int h = e >> 7, d = e & 127;
    kbf[((size_t)h * SEQ + s) * DHEAD + d] = (__bf16)k[(size_t)s * EMB + e];
  }
}

// ---------------- Kernel 2c: V -> bf16 transposed, [NH][DH][S] -------------
__global__ void __launch_bounds__(256)
vconv_kernel(const float* __restrict__ v, __bf16* __restrict__ vt) {
  __shared__ float tile[32][33];
  const int sb = blockIdx.x * 32;                 // seq base
  const int db = blockIdx.y * 32;                 // embed-col base
  const int tx = threadIdx.x & 31, ty = threadIdx.x >> 5;
#pragma unroll
  for (int i = 0; i < 4; ++i)
    tile[ty + i * 8][tx] = v[(size_t)(sb + ty + i * 8) * EMB + db + tx];
  __syncthreads();
#pragma unroll
  for (int i = 0; i < 4; ++i) {
    int dg = db + ty + i * 8;
    int h = dg >> 7, d = dg & 127;
    vt[((size_t)h * DHEAD + d) * SEQ + sb + tx] = (__bf16)tile[tx][ty + i * 8];
  }
}

// ---------------- Kernel 3: flash mLSTM core (bf16 WMMA) -------------------
// One wave per (16-row tile, head). grid = (SEQ/16, NHEAD), block = 32.
__global__ void __launch_bounds__(32)
mlstm_flash_kernel(const float* __restrict__ q, const __bf16* __restrict__ kbf,
                   const __bf16* __restrict__ vt, const float* __restrict__ gb,
                   const float* __restrict__ gmx, const float* __restrict__ mb,
                   float* __restrict__ hbuf) {
  __shared__ float ctile[16 * 36];  // 16x32 C tile, stride 36 to split banks
  const int head = blockIdx.y;
  const int ib = blockIdx.x * 16;
  const int lane = threadIdx.x;
  const int lo = lane & 15;
  const int hi = lane >> 4;
  const float rscale = 0.08838834764831845f;  // 1/sqrt(128)
  const v8f vzero = {0.f, 0.f, 0.f, 0.f, 0.f, 0.f, 0.f, 0.f};

  // Resident Q A-fragments: 16x128 bf16 as 4 x (16x32). A layout:
  // row = lane%16; elem e -> kdim = f*32 + hi*16 + e.
  v16bf qa[4];
  const float* qrow = q + (size_t)(ib + lo) * EMB + head * DHEAD;
#pragma unroll
  for (int f = 0; f < 4; ++f) {
    const float* p0 = qrow + f * 32 + hi * 16;
    float4 a0 = *(const float4*)(p0);
    float4 a1 = *(const float4*)(p0 + 4);
    float4 b0 = *(const float4*)(p0 + 8);
    float4 b1 = *(const float4*)(p0 + 12);
    v16bf t;
    t[0]=(__bf16)a0.x; t[1]=(__bf16)a0.y; t[2]=(__bf16)a0.z; t[3]=(__bf16)a0.w;
    t[4]=(__bf16)a1.x; t[5]=(__bf16)a1.y; t[6]=(__bf16)a1.z; t[7]=(__bf16)a1.w;
    t[8]=(__bf16)b0.x; t[9]=(__bf16)b0.y; t[10]=(__bf16)b0.z; t[11]=(__bf16)b0.w;
    t[12]=(__bf16)b1.x; t[13]=(__bf16)b1.y; t[14]=(__bf16)b1.z; t[15]=(__bf16)b1.w;
    qa[f] = t;
  }

  // Per-lane row state: this lane owns rows (ib + r + hi*8), matching the
  // f32 C/D accumulator layout (row = vgpr + laneHi*8).
  float gmax8[8], em8[8], bacc[8], scale8[8];
#pragma unroll
  for (int r = 0; r < 8; ++r) {
    int row = ib + r + hi * 8;
    gmax8[r] = gmx[head * SEQ + row];
    em8[r] = expf(-mb[head * SEQ + row]);  // exp(-max_log_D) floor
    bacc[r] = 0.f;
  }
#pragma unroll
  for (int r = 0; r < 8; ++r)
    scale8[r] = expf(gmax8[0] - gmax8[r]);  // <=1 (gmax nondecreasing in row)

  v8f acc[8];
#pragma unroll
  for (int c = 0; c < 8; ++c) acc[c] = vzero;

  const __bf16* kh = kbf + (size_t)head * SEQ * DHEAD;
  const __bf16* vh = vt + (size_t)head * DHEAD * SEQ;
  const float* gh = gb + head * SEQ;

  const int full = ib >> 5;            // tiles strictly below the diagonal
  const int nt = (ib + 16 + 31) >> 5;  // total 32-wide causal column tiles
  for (int jt = 0; jt < nt; ++jt) {
    const int jb = jt * 32;
    // ---- S = Q K^T for two 16-col halves; B layout: col n = lane%16,
    //      elem e -> kdim = f*32 + hi*16 + e (16 contiguous bf16 = 32B).
    v8f s0 = vzero, s1 = vzero;
#pragma unroll
    for (int f = 0; f < 4; ++f) {
      v16bf b0 = *(const v16bf*)(kh + (size_t)(jb + lo) * DHEAD + f * 32 + hi * 16);
      s0 = __builtin_amdgcn_wmma_f32_16x16x32_bf16(false, qa[f], false, b0,
                                                   (short)0, s0, false, false);
      v16bf b1 = *(const v16bf*)(kh + (size_t)(jb + 16 + lo) * DHEAD + f * 32 + hi * 16);
      s1 = __builtin_amdgcn_wmma_f32_16x16x32_bf16(false, qa[f], false, b1,
                                                   (short)0, s1, false, false);
    }
    // ---- C = (qk/sqrt(dh)) * exp(g_j - gmax_i), causal-masked.
    const float g0 = gh[jb + lo];
    const float g1 = gh[jb + 16 + lo];
    if (jt < full) {
      // Fast path: whole tile below diagonal for every row of this tile.
      // exp(g - gmax_r) = exp(g - gmax_base) * scale8[r]; both factors <= 1.
      const float e0 = expf(g0 - gmax8[0]) * rscale;
      const float e1 = expf(g1 - gmax8[0]) * rscale;
#pragma unroll
      for (int r = 0; r < 8; ++r) {
        float c0 = s0[r] * (e0 * scale8[r]);
        float c1 = s1[r] * (e1 * scale8[r]);
        bacc[r] += c0 + c1;
        ctile[(r + hi * 8) * 36 + lo] = c0;
        ctile[(r + hi * 8) * 36 + 16 + lo] = c1;
      }
    } else {
      const int j0 = jb + lo, j1 = jb + 16 + lo;
#pragma unroll
      for (int r = 0; r < 8; ++r) {
        const int row = ib + r + hi * 8;
        float p0 = (j0 <= row) ? expf(g0 - gmax8[r]) : 0.f;  // arg <= 0: safe
        float p1 = (j1 <= row) ? expf(g1 - gmax8[r]) : 0.f;
        float c0 = s0[r] * rscale * p0;
        float c1 = s1[r] * rscale * p1;
        bacc[r] += c0 + c1;
        ctile[(r + hi * 8) * 36 + lo] = c0;
        ctile[(r + hi * 8) * 36 + 16 + lo] = c1;
      }
    }
    // ---- Re-read C in A layout (same-wave LDS ops are in order).
    v16bf ca;
    {
      const float* cr = &ctile[lo * 36 + hi * 16];
      float4 a0 = *(const float4*)(cr);
      float4 a1 = *(const float4*)(cr + 4);
      float4 b0 = *(const float4*)(cr + 8);
      float4 b1 = *(const float4*)(cr + 12);
      ca[0]=(__bf16)a0.x; ca[1]=(__bf16)a0.y; ca[2]=(__bf16)a0.z; ca[3]=(__bf16)a0.w;
      ca[4]=(__bf16)a1.x; ca[5]=(__bf16)a1.y; ca[6]=(__bf16)a1.z; ca[7]=(__bf16)a1.w;
      ca[8]=(__bf16)b0.x; ca[9]=(__bf16)b0.y; ca[10]=(__bf16)b0.z; ca[11]=(__bf16)b0.w;
      ca[12]=(__bf16)b1.x; ca[13]=(__bf16)b1.y; ca[14]=(__bf16)b1.z; ca[15]=(__bf16)b1.w;
    }
    // ---- acc += C @ V. B from vt[NH][DH][S]: n = dh col (lane%16),
    //      elem e -> j = jb + hi*16 + e: 16 contiguous bf16 = 32B aligned.
#pragma unroll
    for (int c = 0; c < 8; ++c) {
      v16bf vb = *(const v16bf*)(vh + (size_t)(c * 16 + lo) * SEQ + jb + hi * 16);
      acc[c] = __builtin_amdgcn_wmma_f32_16x16x32_bf16(false, ca, false, vb,
                                                       (short)0, acc[c], false, false);
    }
  }

  // ---- Finalize: reduce row sums across the 16-lane half, normalize, store.
#pragma unroll
  for (int r = 0; r < 8; ++r) {
    float b = bacc[r];
    b += __shfl_xor(b, 1, 32);
    b += __shfl_xor(b, 2, 32);
    b += __shfl_xor(b, 4, 32);
    b += __shfl_xor(b, 8, 32);
    float inv = 1.f / (fmaxf(fabsf(b), em8[r]) + EPS_N);
    const int row = ib + r + hi * 8;
    float* op = hbuf + (size_t)row * EMB + head * DHEAD + lo;
#pragma unroll
    for (int c = 0; c < 8; ++c) op[c * 16] = acc[c][r] * inv;
  }
}

// ---------------- Kernel 4: LayerNorm over E (weight only) -----------------
__global__ void __launch_bounds__(256)
ln_kernel(const float* __restrict__ hbuf, const float* __restrict__ lnw,
          float* __restrict__ out) {
  __shared__ float ssum[8], ssq[8];
  const int row = blockIdx.x;
  const int t = threadIdx.x;
  float x[4];
  float s = 0.f, q2 = 0.f;
#pragma unroll
  for (int i = 0; i < 4; ++i) {
    x[i] = hbuf[(size_t)row * EMB + t * 4 + i];
    s += x[i];
    q2 += x[i] * x[i];
  }
#pragma unroll
  for (int m = 16; m >= 1; m >>= 1) {
    s += __shfl_xor(s, m, 32);
    q2 += __shfl_xor(q2, m, 32);
  }
  if ((t & 31) == 0) { ssum[t >> 5] = s; ssq[t >> 5] = q2; }
  __syncthreads();
  float ts = 0.f, tq = 0.f;
#pragma unroll
  for (int i = 0; i < 8; ++i) { ts += ssum[i]; tq += ssq[i]; }
  const float mu = ts * (1.f / EMB);
  const float var = tq * (1.f / EMB) - mu * mu;
  const float inv = rsqrtf(var + LN_EPS);
#pragma unroll
  for (int i = 0; i < 4; ++i) {
    int c = t * 4 + i;
    out[(size_t)row * EMB + c] = (x[i] - mu) * inv * lnw[c];
  }
}

extern "C" void kernel_launch(void* const* d_in, const int* in_sizes, int n_in,
                              void* d_out, int out_size, void* d_ws, size_t ws_size,
                              hipStream_t stream) {
  const float* q   = (const float*)d_in[0];
  const float* k   = (const float*)d_in[1];
  const float* v   = (const float*)d_in[2];
  const float* igw = (const float*)d_in[3];
  const float* igb = (const float*)d_in[4];
  const float* fgw = (const float*)d_in[5];
  const float* fgb = (const float*)d_in[6];
  const float* lnw = (const float*)d_in[7];

  float* hbuf = (float*)d_ws;                       // [SEQ, EMB] f32 (8 MB)
  __bf16* kbf = (__bf16*)(hbuf + (size_t)SEQ * EMB);        // [NH][S][DH] (4 MB)
  __bf16* vtb = kbf + (size_t)NHEAD * SEQ * DHEAD;          // [NH][DH][S] (4 MB)
  float* gbuf = (float*)(vtb + (size_t)NHEAD * DHEAD * SEQ);// [NH, SEQ] g
  float* gmx  = gbuf + NHEAD * SEQ;                 // [NH, SEQ] prefix-max g
  float* mb   = gmx  + NHEAD * SEQ;                 // [NH, SEQ] m = csum + gmax
  float* igt  = mb   + NHEAD * SEQ;                 // [NH, SEQ] ig preact
  float* lft  = igt  + NHEAD * SEQ;                 // [NH, SEQ] logsigmoid(fg)

  gate_kernel<<<SEQ, 256, 0, stream>>>(q, k, v, igw, igb, fgw, fgb, igt, lft);
  scan_kernel<<<NHEAD, 32, 0, stream>>>(igt, lft, gbuf, gmx, mb);
  kconv_kernel<<<SEQ, 256, 0, stream>>>(k, kbf);
  dim3 tgrid(SEQ / 32, EMB / 32);
  vconv_kernel<<<tgrid, 256, 0, stream>>>(v, vtb);
  dim3 grid(SEQ / 16, NHEAD);
  mlstm_flash_kernel<<<grid, 32, 0, stream>>>(q, kbf, vtb, gbuf, gmx, mb, hbuf);
  ln_kernel<<<SEQ, 256, 0, stream>>>(hbuf, lnw, (float*)d_out);
}
